// IDCST2_1133871366822
// MI455X (gfx1250) — matrix operand order repaired and checked by reading, not compiled, not run on hardware
//
#include <hip/hip_runtime.h>

// ---------------------------------------------------------------------------
// IDCST2 on MI455X (gfx1250): two f16 WMMA GEMMs with f32 accumulation.
//   Stage 1: T = X * S          (store transposed -> T^T, f16)
//   Stage 2: P = T^T * C = Y^T  (store transposed -> Y, f32)
// Data path: global -> LDS via global_load_async_to_lds_b128 (ASYNCcnt,
// double buffered, K-slab 64), LDS -> VGPR fragments via ds_load_b128,
// 16 x v_wmma_f32_16x16x32_f16 per sync round.
// ---------------------------------------------------------------------------

typedef __attribute__((ext_vector_type(16))) _Float16 v16h;
typedef __attribute__((ext_vector_type(8)))  _Float16 v8h;
typedef __attribute__((ext_vector_type(4)))  _Float16 v4h;
typedef __attribute__((ext_vector_type(8)))  float    v8f;
typedef __attribute__((ext_vector_type(4)))  float    v4f;

#define PI_F 3.14159265358979323846f

// ---------------------------------------------------------------- helpers ---

__global__ __launch_bounds__(256)
void f32_to_f16_vec4(const float* __restrict__ in, _Float16* __restrict__ outp, long n4) {
  long i = (long)blockIdx.x * blockDim.x + threadIdx.x;
  if (i < n4) {
    v4f v = ((const v4f*)in)[i];
    v4h h = { (_Float16)v[0], (_Float16)v[1], (_Float16)v[2], (_Float16)v[3] };
    ((v4h*)outp)[i] = h;
  }
}

// St[v*N + q] = sin(pi/N * q * (v + 0.5))   (rows = v, cols = q)
__global__ __launch_bounds__(256)
void gen_sin_T(_Float16* __restrict__ St, int N) {
  long idx = (long)blockIdx.x * blockDim.x + threadIdx.x;
  long total = (long)N * N;
  if (idx < total) {
    int v = (int)(idx / N);
    int q = (int)(idx % N);
    float ang = (PI_F / (float)N) * (float)q * ((float)v + 0.5f);
    St[idx] = (_Float16)sinf(ang);
  }
}

// Ct[u*M + p] = cos(pi/M * p * (u + 0.5))   (rows = u, cols = p)
__global__ __launch_bounds__(256)
void gen_cos_T(_Float16* __restrict__ Ct, int M) {
  long idx = (long)blockIdx.x * blockDim.x + threadIdx.x;
  long total = (long)M * M;
  if (idx < total) {
    int u = (int)(idx / M);
    int p = (int)(idx % M);
    float ang = (PI_F / (float)M) * (float)p * ((float)u + 0.5f);
    Ct[idx] = (_Float16)cosf(ang);
  }
}

// ----------------------------------------------------- CDNA5 async helpers ---

// LDS byte offset of a generic pointer that points into __shared__ memory.
__device__ __forceinline__ unsigned lds_off(const _Float16* p) {
  return (unsigned)(unsigned long long)
         (const __attribute__((address_space(3))) _Float16*)p;
}

// global -> LDS async copy, 16 bytes per lane (tracked by ASYNCcnt).
__device__ __forceinline__ void async_copy_b128(unsigned ldsByteOff,
                                                const _Float16* gsrc) {
  asm volatile("global_load_async_to_lds_b128 %0, %1, off"
               :: "v"(ldsByteOff), "v"(gsrc)
               : "memory");
}

__device__ __forceinline__ void wait_async0() {
  asm volatile("s_wait_asynccnt 0x0" ::: "memory");
}

// ------------------------------------------------------------- WMMA GEMM ---
// D(MxN) = A(MxK, row-major) * B(KxN) with B supplied TRANSPOSED
// (Bt, NxK row-major). Output stored TRANSPOSED: Out[n*ldo + m] = D[m,n]
// (per the CDNA5 C/D VGPR layout this makes each lane's 8 acc entries 8
//  consecutive output elements -> b128 stores).
//
// Block: 256 threads = 8 wave32s; block tile 128(M) x 128(N);
// wave tile 64(M) x 32(N) = 4x2 grid of 16x16x32 WMMAs; K slab 64
// (two 32-deep WMMA k-steps per sync round), double-buffered LDS (72 KB).

constexpr int BM = 128, BN = 128;
constexpr int KSLAB = 64;                  // K depth staged per sync round
constexpr int LDS_STRIDE = KSLAB + 8;      // 72 halves (144 B) rows: bank spread

template <typename OutT>
__global__ __launch_bounds__(256)
void wmma_gemm_tn(const _Float16* __restrict__ A,  int lda, long strideA,
                  const _Float16* __restrict__ Bt, int ldb, long strideB,
                  OutT* __restrict__ Out,          int ldo, long strideO,
                  int K)
{
  __shared__ _Float16 sA[2][BM * LDS_STRIDE];   // 2 x 18 KB
  __shared__ _Float16 sB[2][BN * LDS_STRIDE];   // 2 x 18 KB

  const int batch = blockIdx.z;
  A   += (long)batch * strideA;
  Bt  += (long)batch * strideB;
  Out += (long)batch * strideO;

  const int tid  = threadIdx.x;
  const int wave = tid >> 5;
  const int lane = tid & 31;
  const int l16  = lane & 15;
  const int hf   = lane >> 4;          // lanes 0-15 / 16-31

  const int wm = wave & 1;             // 2 waves along M
  const int wn = wave >> 1;            // 4 waves along N
  const int mBase = blockIdx.y * BM + wm * 64;
  const int nBase = blockIdx.x * BN + wn * 32;

  // Block-local slab bases in global memory.
  const _Float16* gA = A  + (long)(blockIdx.y * BM) * lda;
  const _Float16* gB = Bt + (long)(blockIdx.x * BN) * ldb;

  // Copy-chunk mapping: slab = 128 rows x 128 B; 1024 chunks of 16 B;
  // each thread moves chunks {tid, tid+256, tid+512, tid+768} per matrix.
  const int crow = tid >> 3;           // 0..31
  const int ckc  = (tid & 7) * 8;      // half offset within slab: 0..56

  auto issue_slab = [&](int bi, int kb) {
#pragma unroll
    for (int j = 0; j < 4; ++j) {
      const int row = crow + j * 32;
      async_copy_b128(lds_off(&sA[bi][row * LDS_STRIDE + ckc]),
                      gA + (long)row * lda + kb + ckc);
      async_copy_b128(lds_off(&sB[bi][row * LDS_STRIDE + ckc]),
                      gB + (long)row * ldb + kb + ckc);
    }
  };

  v8f acc[4][2] = {};                  // [m-tile][n-tile], 16x16 f32 each

  issue_slab(0, 0);
  wait_async0();
  __syncthreads();

  int buf = 0;
  for (int kb = 0; kb < K; kb += KSLAB) {
    if (kb + KSLAB < K) {
      issue_slab(buf ^ 1, kb + KSLAB);
      if (kb + 2 * KSLAB < K) {        // HBM->L2 lookahead for slab after next
        __builtin_prefetch((const void*)(gA + (long)crow * lda + kb + 2 * KSLAB), 0, 1);
        __builtin_prefetch((const void*)(gB + (long)crow * ldb + kb + 2 * KSLAB), 0, 1);
      }
    }

    const _Float16* a0 = &sA[buf][0];
    const _Float16* b0 = &sB[buf][0];

    // Two 32-deep WMMA k-steps per slab.
#pragma unroll
    for (int ks = 0; ks < KSLAB; ks += 32) {
      // A fragments (16-bit A layout: half 0 -> K 0..7 & 16..23,
      //                               half 1 -> K 8..15 & 24..31)
      v16h afr[4];
#pragma unroll
      for (int mt = 0; mt < 4; ++mt) {
        const _Float16* ap =
            a0 + (wm * 64 + mt * 16 + l16) * LDS_STRIDE + ks + hf * 8;
        v8h lo = *(const v8h*)ap;
        v8h hi = *(const v8h*)(ap + 16);
        afr[mt] = __builtin_shufflevector(lo, hi,
                   0, 1, 2, 3, 4, 5, 6, 7, 8, 9, 10, 11, 12, 13, 14, 15);
      }

      // B fragments (lane owns column n = l16; half 0 -> K 0..7 & 16..23,
      //              half 1 -> K 8..15 & 24..31)
      v16h bfr[2];
#pragma unroll
      for (int nt = 0; nt < 2; ++nt) {
        const _Float16* bp =
            b0 + (wn * 32 + nt * 16 + l16) * LDS_STRIDE + ks + hf * 8;
        v8h lo = *(const v8h*)bp;
        v8h hi = *(const v8h*)(bp + 16);
        bfr[nt] = __builtin_shufflevector(lo, hi,
                   0, 1, 2, 3, 4, 5, 6, 7, 8, 9, 10, 11, 12, 13, 14, 15);
      }

      // 8 WMMAs: D = A x B + C
#pragma unroll
      for (int mt = 0; mt < 4; ++mt) {
#pragma unroll
        for (int nt = 0; nt < 2; ++nt) {
          acc[mt][nt] = __builtin_amdgcn_wmma_f32_16x16x32_f16(
              false, afr[mt], false, bfr[nt],
              (short)0, acc[mt][nt], false, false);
        }
      }
    }

    wait_async0();       // own async loads for next slab have landed
    __syncthreads();     // everyone's slab writes visible; prev reads done
    buf ^= 1;
  }

  // Transposed store: D element (row = r + 8*hf, col = l16) ->
  // Out[(nBase + col)*ldo + mBase + row]; 8 acc entries per lane are
  // 8 consecutive output elements.
#pragma unroll
  for (int nt = 0; nt < 2; ++nt) {
#pragma unroll
    for (int mt = 0; mt < 4; ++mt) {
      OutT* op = Out + (long)(nBase + nt * 16 + l16) * ldo
                     + mBase + mt * 16 + 8 * hf;
      if constexpr (sizeof(OutT) == 4) {   // f32 final output
        v4f x0 = { acc[mt][nt][0], acc[mt][nt][1], acc[mt][nt][2], acc[mt][nt][3] };
        v4f x1 = { acc[mt][nt][4], acc[mt][nt][5], acc[mt][nt][6], acc[mt][nt][7] };
        *(v4f*)(op)     = x0;
        *(v4f*)(op + 4) = x1;
      } else {                             // f16 intermediate
        v8h h;
#pragma unroll
        for (int r = 0; r < 8; ++r) h[r] = (_Float16)acc[mt][nt][r];
        *(v8h*)op = h;
      }
    }
  }
}

// ------------------------------------------------------------------ launch ---

extern "C" void kernel_launch(void* const* d_in, const int* in_sizes, int n_in,
                              void* d_out, int out_size, void* d_ws, size_t ws_size,
                              hipStream_t stream) {
  (void)in_sizes; (void)n_in; (void)out_size; (void)ws_size;

  constexpr int  B = 4, M = 2048, N = 2048;
  constexpr long MN = (long)M * N;

  const float* x   = (const float*)d_in[0];
  float*       out = (float*)d_out;

  char* ws = (char*)d_ws;
  _Float16* Xh = (_Float16*)(ws);                                   // B*M*N f16 (32 MB)
  _Float16* St = (_Float16*)(ws + (long)B * MN * 2);                // N*N  f16 ( 8 MB)
  _Float16* Ct = (_Float16*)(ws + (long)B * MN * 2 + MN * 2);       // M*M  f16 ( 8 MB)
  _Float16* Tt = (_Float16*)(ws + (long)B * MN * 2 + 2 * MN * 2);   // B*N*M f16 (32 MB)

  // 1) convert x -> f16 (vectorized 4-wide)
  {
    long n4 = (long)B * MN / 4;
    f32_to_f16_vec4<<<(unsigned)((n4 + 255) / 256), 256, 0, stream>>>(x, Xh, n4);
  }
  // 2) generate basis matrices (transposed layouts)
  gen_sin_T<<<(unsigned)(((long)N * N + 255) / 256), 256, 0, stream>>>(St, N);
  gen_cos_T<<<(unsigned)(((long)M * M + 255) / 256), 256, 0, stream>>>(Ct, M);

  dim3 block(256, 1, 1);

  // 3) Stage 1: T = Xh * S  -> store transposed => Tt = T^T  [v, p], f16
  //    A = Xh [p, q] (row-major), Bt = St [v, q] (= S^T), M-dim = p, N-dim = v.
  {
    dim3 grid(N / BN, M / BM, B);
    wmma_gemm_tn<_Float16><<<grid, block, 0, stream>>>(
        Xh, N, MN,        // A, lda, batch stride
        St, N, 0,         // Bt shared across batch
        Tt, M, MN,        // Out = T^T, ldo = M
        N);               // K = q-dim
  }

  // 4) Stage 2: P = Tt * C = Y^T -> store transposed => Y [u, v], f32
  //    A = Tt [v, p], Bt = Ct [u, p] (= C^T), M-dim = v, N-dim = u.
  {
    dim3 grid(M / BN, N / BM, B);
    wmma_gemm_tn<float><<<grid, block, 0, stream>>>(
        Tt, M, MN,        // A, lda, batch stride
        Ct, M, 0,         // Bt shared across batch
        out, N, MN,       // Out = Y, ldo = N
        M);               // K = p-dim
  }
}
